// MessageLayer_11862699671910
// MI455X (gfx1250) — matrix operand order repaired and checked by reading, not compiled, not run on hardware
//
#include <hip/hip_runtime.h>
#include <hip/hip_bf16.h>

typedef __attribute__((ext_vector_type(16))) _Float16 v16h;
typedef __attribute__((ext_vector_type(8)))  _Float16 v8h;
typedef __attribute__((ext_vector_type(8)))  float    v8f;

#define HID 32
#define TDIM 128

__device__ __forceinline__ float sigf(float v) { return 1.0f / (1.0f + __expf(-v)); }
__device__ __forceinline__ float siluf(float v) { return v * sigf(v); }

__device__ __forceinline__ v8f wmma16(v16h a, v16h b, v8f c) {
    return __builtin_amdgcn_wmma_f32_16x16x32_f16(false, a, false, b, (short)0, c, false, false);
}

__device__ __forceinline__ v16h pack16(v8h lo, v8h hi) {
    v16h r;
#pragma unroll
    for (int e = 0; e < 8; ++e) { r[e] = lo[e]; r[e + 8] = hi[e]; }
    return r;
}

// A-fragment (16x32 f16): lane holds row (lane&15); halves e=0..7 -> K=koff+e,
// e=8..15 -> K=16+koff+(e-8), koff = (lane>=16)?8:0.
__device__ __forceinline__ v16h load_afrag_f16(const _Float16* rowbase, int koff) {
    v8h lo = *(const v8h*)(rowbase + koff);
    v8h hi = *(const v8h*)(rowbase + 16 + koff);
    return pack16(lo, hi);
}

__device__ __forceinline__ v16h load_afrag_f32(const float* rowbase, int koff) {
    v16h r;
#pragma unroll
    for (int e = 0; e < 8; ++e) {
        r[e]     = (_Float16)rowbase[koff + e];
        r[e + 8] = (_Float16)rowbase[16 + koff + e];
    }
    return r;
}

// B-fragment from pre-swizzled buffer: frag = kc*NH + nh; per-lane 16 contiguous halves.
__device__ __forceinline__ v16h load_bfrag(const _Float16* swz, int frag, int lane) {
    return *(const v16h*)(swz + ((frag * 32 + lane) << 4));
}

// ---------------- weight swizzle: W (K x Nout) f32 -> WMMA B-layout f16 ----------------
__global__ void swz_kernel(const float* __restrict__ W, _Float16* __restrict__ dst,
                           int K, int Nout) {
    int t = blockIdx.x * blockDim.x + threadIdx.x;
    int total = K * Nout;
    if (t >= total) return;
    int e = t & 15;
    int lane = (t >> 4) & 31;
    int frag = t >> 9;
    int NH = Nout >> 4;
    int kc = frag / NH, nh = frag - kc * NH;
    int k = kc * 32 + ((lane >= 16) ? 8 : 0) + (e & 7) + ((e >= 8) ? 16 : 0);
    int n = nh * 16 + (lane & 15);
    dst[t] = (_Float16)W[k * Nout + n];
}

// ---------------- zero scratch ----------------
__global__ void zero_kernel(float* __restrict__ p, int n) {
    int t = blockIdx.x * blockDim.x + threadIdx.x;
    if (t < n) p[t] = 0.0f;
}

// ---------------- time MLP: ss[g] = silu(time@W1+b1)@W2+b2 ----------------
__global__ __launch_bounds__(64) void time_kernel(
    const float* __restrict__ time, const float* __restrict__ w1, const float* __restrict__ b1,
    const float* __restrict__ w2, const float* __restrict__ b2, float* __restrict__ ss) {
    __shared__ float hbuf[64];
    int g = blockIdx.x, t = threadIdx.x;
    float acc = b1[t];
    for (int k = 0; k < TDIM; ++k) acc += time[g * TDIM + k] * w1[k * 64 + t];
    hbuf[t] = siluf(acc);
    __syncthreads();
    float acc2 = b2[t];
    for (int k = 0; k < 64; ++k) acc2 += hbuf[k] * w2[k * 64 + t];
    ss[g * 64 + t] = acc2;
}

// ---------------- modulation: x_mod = silu(x*(1+scale)+shift), f32 + f16 copies ----------------
__global__ void mod_kernel(const float* __restrict__ x, const int* __restrict__ batch,
                           const float* __restrict__ ss, float* __restrict__ xm32,
                           _Float16* __restrict__ xm16, int N) {
    int t = blockIdx.x * blockDim.x + threadIdx.x;
    if (t >= N * HID) return;
    int node = t >> 5, f = t & 31;
    int b = batch[node];
    float sc = ss[b * 64 + f], sh = ss[b * 64 + 32 + f];
    float v = siluf(x[t] * (1.0f + sc) + sh);
    xm32[t] = v;
    xm16[t] = (_Float16)v;
}

// ---------------- edge kernel: one wave per 16 edges, WMMA everywhere ----------------
__global__ __launch_bounds__(256) void edge_kernel(
    const int* __restrict__ ei, const float* __restrict__ pos,
    const _Float16* __restrict__ xm16,
    const float* __restrict__ rbf_means, const float* __restrict__ rbf_betas,
    const _Float16* __restrict__ wds, const _Float16* __restrict__ w1s,
    const float* __restrict__ msg_b1, const _Float16* __restrict__ w2s,
    const float* __restrict__ msg_b2, const _Float16* __restrict__ gws,
    const float* __restrict__ gate_b, const _Float16* __restrict__ cw1s,
    const float* __restrict__ coord_b1, const float* __restrict__ coord_w2,
    const float* __restrict__ coord_b2,
    float* __restrict__ m_i, float* __restrict__ sum_p, float* __restrict__ cnt,
    int E) {
    __shared__ __align__(16) _Float16 lds[8 * 512];
    int lane = threadIdx.x & 31;
    int wv = (blockIdx.x * blockDim.x + threadIdx.x) >> 5;
    int tile = wv * 16;
    if (tile >= E) return;
    _Float16* L = lds + (threadIdx.x >> 5) * 512;
    int mrow = lane & 15;
    int half = lane >> 4;
    int koff = half * 8;
    int nlo = lane & 15;

    int er = min(tile + mrow, E - 1);
    int vi = ei[er];
    int vj = ei[E + er];

    float dx = pos[vi * 3 + 0] - pos[vj * 3 + 0];
    float dy = pos[vi * 3 + 1] - pos[vj * 3 + 1];
    float dz = pos[vi * 3 + 2] - pos[vj * 3 + 2];
    float dist = sqrtf(dx * dx + dy * dy + dz * dz);
    float dcl = fminf(dist, 5.0f);
    float cutoff = 0.5f * (__cosf(dcl * (3.14159265358979323846f / 5.0f)) + 1.0f);
    float expd = __expf(-dist);

    // rbf A-fragment built directly in WMMA A layout
    v16h arbf;
#pragma unroll
    for (int e = 0; e < 16; ++e) {
        int k = (e < 8) ? (koff + e) : (16 + koff + (e - 8));
        float d = expd - rbf_means[k];
        arbf[e] = (_Float16)(cutoff * __expf(-rbf_betas[k] * d * d));
    }

    v16h ai = load_afrag_f16(xm16 + vi * HID, koff);
    v16h aj = load_afrag_f16(xm16 + vj * HID, koff);

    // h = silu([x_i,x_j] @ msg_w1 + b1)   (D layout)
    float hv[2][8];
#pragma unroll
    for (int nh = 0; nh < 2; ++nh) {
        v8f acc = {};
        acc = wmma16(ai, load_bfrag(w1s, 0 * 2 + nh, lane), acc);
        acc = wmma16(aj, load_bfrag(w1s, 1 * 2 + nh, lane), acc);
        float bb = msg_b1[nlo + 16 * nh];
#pragma unroll
        for (int r = 0; r < 8; ++r) hv[nh][r] = siluf(acc[r] + bb);
    }
    // D layout -> LDS [M][n] (f16)
#pragma unroll
    for (int nh = 0; nh < 2; ++nh)
#pragma unroll
        for (int r = 0; r < 8; ++r)
            L[(r + half * 8) * 32 + nlo + 16 * nh] = (_Float16)hv[nh][r];
    __builtin_amdgcn_wave_barrier();   // order cross-lane LDS store -> load

    // gate = sigmoid(x_j @ gate_w + b); demb = (cutoff*rbf) @ w_dist
    float gv[2][8], dev[2][8];
#pragma unroll
    for (int nh = 0; nh < 2; ++nh) {
        v8f ag = {}, ad = {};
        ag = wmma16(aj, load_bfrag(gws, nh, lane), ag);
        ad = wmma16(arbf, load_bfrag(wds, nh, lane), ad);
        float gb = gate_b[nlo + 16 * nh];
#pragma unroll
        for (int r = 0; r < 8; ++r) { gv[nh][r] = sigf(ag[r] + gb); dev[nh][r] = ad[r]; }
    }

    // m_ij = silu(h @ msg_w2 + b2) * demb * gate
    v16h ah = load_afrag_f16(L + mrow * 32, koff);
    __builtin_amdgcn_wave_barrier();   // finish reads of h before overwriting L
    float mv[2][8];
#pragma unroll
    for (int nh = 0; nh < 2; ++nh) {
        v8f acc = {};
        acc = wmma16(ah, load_bfrag(w2s, nh, lane), acc);
        float bb = msg_b2[nlo + 16 * nh];
#pragma unroll
        for (int r = 0; r < 8; ++r)
            mv[nh][r] = siluf(acc[r] + bb) * dev[nh][r] * gv[nh][r];
    }

    // scatter m_ij into m_i[j], and stage f16 copy for coord GEMM
#pragma unroll
    for (int nh = 0; nh < 2; ++nh)
#pragma unroll
        for (int r = 0; r < 8; ++r) {
            int M = r + half * 8;
            int je = __shfl(vj, M, 32);
            L[M * 32 + nlo + 16 * nh] = (_Float16)mv[nh][r];
            if (tile + M < E)
                atomicAdd(&m_i[je * HID + nlo + 16 * nh], mv[nh][r]);
        }
    __builtin_amdgcn_wave_barrier();   // order cross-lane LDS store -> load

    // s = silu(m_ij @ coord_w1 + b) @ coord_w2 + b2  (rank-1 via shuffle reduce)
    v16h am = load_afrag_f16(L + mrow * 32, koff);
    float part[8];
    {
        v8f a0 = {}, a1 = {};
        a0 = wmma16(am, load_bfrag(cw1s, 0, lane), a0);
        a1 = wmma16(am, load_bfrag(cw1s, 1, lane), a1);
        float b0 = coord_b1[nlo], b1v = coord_b1[nlo + 16];
        float c0 = coord_w2[nlo], c1 = coord_w2[nlo + 16];
#pragma unroll
        for (int r = 0; r < 8; ++r)
            part[r] = siluf(a0[r] + b0) * c0 + siluf(a1[r] + b1v) * c1;
    }
#pragma unroll
    for (int r = 0; r < 8; ++r) {
        float p = part[r];
        p += __shfl_xor(p, 1, 32);
        p += __shfl_xor(p, 2, 32);
        p += __shfl_xor(p, 4, 32);
        p += __shfl_xor(p, 8, 32);
        part[r] = p;
    }
    // shuffles for pos scatter done with all lanes active
    int Msrc = (lane & 7) + half * 8;   // lanes 0-7 -> rows 0-7, 16-23 -> rows 8-15
    float sdx = __shfl(dx, Msrc, 32);
    float sdy = __shfl(dy, Msrc, 32);
    float sdz = __shfl(dz, Msrc, 32);
    int js = __shfl(vj, Msrc, 32);
    float pv = 0.0f;
    int r7 = lane & 7;
#pragma unroll
    for (int r = 0; r < 8; ++r) if (r7 == r) pv = part[r];
    float sval = pv + coord_b2[0];
    if ((lane & 15) < 8 && (tile + Msrc) < E) {
        atomicAdd(&sum_p[js * 3 + 0], sdx * sval);
        atomicAdd(&sum_p[js * 3 + 1], sdy * sval);
        atomicAdd(&sum_p[js * 3 + 2], sdz * sval);
        atomicAdd(&cnt[js], 1.0f);
    }
}

// ---------------- node combine: x_new = silu(x + (silu([x,m_i]@W1+b1)@W2+b2)) ----------------
__global__ __launch_bounds__(256) void node_kernel(
    const _Float16* __restrict__ xm16, const float* __restrict__ xm32,
    const float* __restrict__ m_i,
    const _Float16* __restrict__ cb1s, const float* __restrict__ comb_b1,
    const _Float16* __restrict__ cb2s, const float* __restrict__ comb_b2,
    float* __restrict__ out_x, int N) {
    __shared__ __align__(16) _Float16 lds[8 * 512];
    int lane = threadIdx.x & 31;
    int wv = (blockIdx.x * blockDim.x + threadIdx.x) >> 5;
    int tile = wv * 16;
    if (tile >= N) return;
    _Float16* L = lds + (threadIdx.x >> 5) * 512;
    int mrow = lane & 15;
    int half = lane >> 4;
    int koff = half * 8;
    int nlo = lane & 15;

    int node = min(tile + mrow, N - 1);
    v16h a1 = load_afrag_f16(xm16 + node * HID, koff);
    v16h a2 = load_afrag_f32(m_i + node * HID, koff);

#pragma unroll
    for (int nh = 0; nh < 2; ++nh) {
        v8f acc = {};
        acc = wmma16(a1, load_bfrag(cb1s, 0 * 2 + nh, lane), acc);
        acc = wmma16(a2, load_bfrag(cb1s, 1 * 2 + nh, lane), acc);
        float bb = comb_b1[nlo + 16 * nh];
#pragma unroll
        for (int r = 0; r < 8; ++r)
            L[(r + half * 8) * 32 + nlo + 16 * nh] = (_Float16)siluf(acc[r] + bb);
    }
    __builtin_amdgcn_wave_barrier();   // order cross-lane LDS store -> load
    v16h ah = load_afrag_f16(L + mrow * 32, koff);
#pragma unroll
    for (int nh = 0; nh < 2; ++nh) {
        v8f acc = {};
        acc = wmma16(ah, load_bfrag(cb2s, nh, lane), acc);
        float bb = comb_b2[nlo + 16 * nh];
#pragma unroll
        for (int r = 0; r < 8; ++r) {
            int M = r + half * 8;
            int nd = tile + M;
            if (nd < N) {
                int idx = nd * HID + nlo + 16 * nh;
                float v = xm32[idx] + acc[r] + bb;
                out_x[idx] = siluf(v);
            }
        }
    }
}

// ---------------- pos update ----------------
__global__ void pos_kernel(const float* __restrict__ pos, const float* __restrict__ sum_p,
                           const float* __restrict__ cnt, float* __restrict__ out_pos, int N) {
    int t = blockIdx.x * blockDim.x + threadIdx.x;
    if (t >= N) return;
    float c = fmaxf(cnt[t], 1.0f);
    out_pos[t * 3 + 0] = pos[t * 3 + 0] + sum_p[t * 3 + 0] / c;
    out_pos[t * 3 + 1] = pos[t * 3 + 1] + sum_p[t * 3 + 1] / c;
    out_pos[t * 3 + 2] = pos[t * 3 + 2] + sum_p[t * 3 + 2] / c;
}

static inline size_t alignup(size_t v) { return (v + 255) & ~(size_t)255; }

extern "C" void kernel_launch(void* const* d_in, const int* in_sizes, int n_in,
                              void* d_out, int out_size, void* d_ws, size_t ws_size,
                              hipStream_t stream) {
    const float* x        = (const float*)d_in[0];
    const int*   ei       = (const int*)d_in[1];
    const float* pos      = (const float*)d_in[2];
    const float* timep    = (const float*)d_in[3];
    const int*   batch    = (const int*)d_in[4];
    const float* rbf_m    = (const float*)d_in[5];
    const float* rbf_b    = (const float*)d_in[6];
    const float* w_dist   = (const float*)d_in[7];
    const float* msg_w1   = (const float*)d_in[8];
    const float* msg_b1   = (const float*)d_in[9];
    const float* msg_w2   = (const float*)d_in[10];
    const float* msg_b2   = (const float*)d_in[11];
    const float* gate_w   = (const float*)d_in[12];
    const float* gate_b   = (const float*)d_in[13];
    const float* time_w1  = (const float*)d_in[14];
    const float* time_b1  = (const float*)d_in[15];
    const float* time_w2  = (const float*)d_in[16];
    const float* time_b2  = (const float*)d_in[17];
    const float* comb_w1  = (const float*)d_in[18];
    const float* comb_b1  = (const float*)d_in[19];
    const float* comb_w2  = (const float*)d_in[20];
    const float* comb_b2  = (const float*)d_in[21];
    const float* coord_w1 = (const float*)d_in[22];
    const float* coord_b1 = (const float*)d_in[23];
    const float* coord_w2 = (const float*)d_in[24];
    const float* coord_b2 = (const float*)d_in[25];

    int N = in_sizes[0] / HID;
    int E = in_sizes[1] / 2;
    int G = in_sizes[3] / TDIM;

    char* ws = (char*)d_ws;
    size_t off = 0;
    float* ss = (float*)(ws + off);           off = alignup(off + (size_t)G * 64 * 4);
    float* xm32 = (float*)(ws + off);         off = alignup(off + (size_t)N * HID * 4);
    _Float16* xm16 = (_Float16*)(ws + off);   off = alignup(off + (size_t)N * HID * 2);
    float* m_i = (float*)(ws + off);          // m_i, sum_p, cnt contiguous: N*36 floats
    float* sum_p = m_i + (size_t)N * HID;
    float* cnt = sum_p + (size_t)N * 3;       off = alignup(off + (size_t)N * 36 * 4);
    _Float16* w1s  = (_Float16*)(ws + off);   off = alignup(off + 2048 * 2);
    _Float16* w2s  = (_Float16*)(ws + off);   off = alignup(off + 1024 * 2);
    _Float16* gws  = (_Float16*)(ws + off);   off = alignup(off + 1024 * 2);
    _Float16* wds  = (_Float16*)(ws + off);   off = alignup(off + 1024 * 2);
    _Float16* cw1s = (_Float16*)(ws + off);   off = alignup(off + 1024 * 2);
    _Float16* cb1s = (_Float16*)(ws + off);   off = alignup(off + 2048 * 2);
    _Float16* cb2s = (_Float16*)(ws + off);   off = alignup(off + 1024 * 2);
    (void)ws_size; (void)n_in;

    // zero accumulators
    {
        int total = N * 36;
        zero_kernel<<<(total + 255) / 256, 256, 0, stream>>>(m_i, total);
    }
    // weight swizzles (f32 -> f16 WMMA B layout)
    swz_kernel<<<(2048 + 255) / 256, 256, 0, stream>>>(msg_w1, w1s, 64, HID);
    swz_kernel<<<(1024 + 255) / 256, 256, 0, stream>>>(msg_w2, w2s, HID, HID);
    swz_kernel<<<(1024 + 255) / 256, 256, 0, stream>>>(gate_w, gws, HID, HID);
    swz_kernel<<<(1024 + 255) / 256, 256, 0, stream>>>(w_dist, wds, HID, HID);
    swz_kernel<<<(1024 + 255) / 256, 256, 0, stream>>>(coord_w1, cw1s, HID, HID);
    swz_kernel<<<(2048 + 255) / 256, 256, 0, stream>>>(comb_w1, cb1s, 64, HID);
    swz_kernel<<<(1024 + 255) / 256, 256, 0, stream>>>(comb_w2, cb2s, HID, HID);

    // time MLP
    time_kernel<<<G, 64, 0, stream>>>(timep, time_w1, time_b1, time_w2, time_b2, ss);

    // node modulation
    mod_kernel<<<(N * HID + 255) / 256, 256, 0, stream>>>(x, batch, ss, xm32, xm16, N);

    // per-edge WMMA pipeline
    {
        int waves = (E + 15) / 16;
        int blocks = (waves + 7) / 8;
        edge_kernel<<<blocks, 256, 0, stream>>>(ei, pos, xm16, rbf_m, rbf_b,
                                                wds, w1s, msg_b1, w2s, msg_b2,
                                                gws, gate_b, cw1s, coord_b1,
                                                coord_w2, coord_b2,
                                                m_i, sum_p, cnt, E);
    }

    float* out_x = (float*)d_out;
    float* out_pos = out_x + (size_t)N * HID;
    (void)out_size;

    // per-node combine (WMMA)
    {
        int waves = (N + 15) / 16;
        int blocks = (waves + 7) / 8;
        node_kernel<<<blocks, 256, 0, stream>>>(xm16, xm32, m_i, cb1s, comb_b1,
                                                cb2s, comb_b2, out_x, N);
    }
    // pos update
    pos_kernel<<<(N + 255) / 256, 256, 0, stream>>>(pos, sum_p, cnt, out_pos, N);
}
